// BitNetLinearNative_87471303951027
// MI455X (gfx1250) — 8-bit matrix operand order repaired and checked by reading, compiled, not run on hardware
//
#include <hip/hip_runtime.h>
#include <hip/hip_bf16.h>

#define IN_F   8192
#define OUT_F  28672
#define BATCH  8
#define KBLK   (IN_F / 64)
#define I4     (IN_F / 4)

typedef int v8i __attribute__((ext_vector_type(8)));

__device__ __forceinline__ int unpack4(int p) {
    unsigned t0 =  (unsigned)p        & 3u;
    unsigned t1 = ((unsigned)p >> 2)  & 3u;
    unsigned t2 = ((unsigned)p >> 4)  & 3u;
    unsigned t3 = ((unsigned)p >> 6)  & 3u;
    unsigned s  = t0 | (t1 << 8) | (t2 << 16) | (t3 << 24);
    return (int)(((s | 0x80808080u) - 0x01010101u) ^ 0x80808080u);
}

__global__ void init_ws(unsigned* amax) { amax[0] = 0u; }

__global__ void absmax_kernel(const float* __restrict__ x, unsigned* __restrict__ amax) {
    __shared__ float sm[256];
    int tid = blockIdx.x * 256 + threadIdx.x;
    float m = 0.f;
    for (int i = tid; i < BATCH * IN_F; i += gridDim.x * 256) m = fmaxf(m, fabsf(x[i]));
    sm[threadIdx.x] = m;
    __syncthreads();
    for (int off = 128; off > 0; off >>= 1) {
        if (threadIdx.x < off) sm[threadIdx.x] = fmaxf(sm[threadIdx.x], sm[threadIdx.x + off]);
        __syncthreads();
    }
    if (threadIdx.x == 0) atomicMax(amax, __float_as_uint(sm[0]));
}

__global__ void quantize_kernel(const float* __restrict__ x,
                                const unsigned* __restrict__ amax,
                                int* __restrict__ aws) {
    int tid  = blockIdx.x * 256 + threadIdx.x;
    int kb   = tid >> 8;
    int rem  = tid & 255;
    int lane = rem >> 3;
    int v    = rem & 7;
    int M    = lane & 15;
    int half = lane >> 4;
    int K0   = kb * 64 + ((v >> 2) << 5) + (half << 4) + ((v & 3) << 2);
    unsigned d = 0u;
    if (M < BATCH) {
        float am  = fmaxf(__uint_as_float(amax[0]), 1e-5f);
        float inv = 127.f / am;
        const float* xp = x + M * IN_F + K0;
        #pragma unroll
        for (int b = 0; b < 4; ++b) {
            int q = (int)rintf(xp[b] * inv);
            q = q > 127 ? 127 : (q < -128 ? -128 : q);
            d |= ((unsigned)q & 0xFFu) << (8 * b);
        }
    }
    aws[tid] = (int)d;
}

__global__ void __launch_bounds__(256) gemm_kernel(
        const int* __restrict__ packed, const int* __restrict__ aws,
        const unsigned* __restrict__ amax, const float* __restrict__ wscale,
        const float* __restrict__ bias, float* __restrict__ out) {
    const int lane = threadIdx.x & 31;
    const int wave = (blockIdx.x << 3) + (threadIdx.x >> 5);
    const int n0   = wave << 4;
    const int half = lane >> 4;
    const int row  = n0 + (lane & 15);

    const int4* __restrict__ W4 = (const int4*)packed;
    const int4* __restrict__ A4 = (const int4*)aws;
    const long rb = (long)row * (I4 / 4);

    v8i acc = {};
    #pragma unroll 2
    for (int kb = 0; kb < KBLK; ++kb) {
        int4 pa0 = A4[kb * 64 + lane * 2];
        int4 pa1 = A4[kb * 64 + lane * 2 + 1];
        int4 pb0 = W4[rb + kb * 4 + half];
        int4 pb1 = W4[rb + kb * 4 + half + 2];
        __builtin_prefetch(&W4[rb + (kb + 16) * 4 + half], 0, 0);

        v8i a, b;
        a[0] = pa0.x; a[1] = pa0.y; a[2] = pa0.z; a[3] = pa0.w;
        a[4] = pa1.x; a[5] = pa1.y; a[6] = pa1.z; a[7] = pa1.w;
        b[0] = unpack4(pb0.x); b[1] = unpack4(pb0.y);
        b[2] = unpack4(pb0.z); b[3] = unpack4(pb0.w);
        b[4] = unpack4(pb1.x); b[5] = unpack4(pb1.y);
        b[6] = unpack4(pb1.z); b[7] = unpack4(pb1.w);

        acc = __builtin_amdgcn_wmma_i32_16x16x64_iu8(true, a, true, b, acc, false, false);
    }

    float am = fmaxf(__uint_as_float(amax[0]), 1e-5f);
    float s  = wscale[0] * (am / 127.f);
    if (lane < 16) {
        int n = n0 + lane;
        float bsv = bias[n];
        #pragma unroll
        for (int r = 0; r < BATCH; ++r)
            out[r * OUT_F + n] = (float)acc[r] * s + bsv;
    }
}

extern "C" void kernel_launch(void* const* d_in, const int* in_sizes, int n_in,
                              void* d_out, int out_size, void* d_ws, size_t ws_size,
                              hipStream_t stream) {
    const float* x      = (const float*)d_in[0];
    const int*   pw     = (const int*)d_in[1];
    const float* wscale = (const float*)d_in[2];
    const float* bias   = (const float*)d_in[3];
    float*       out    = (float*)d_out;

    unsigned* amax = (unsigned*)d_ws;
    int*      aws  = (int*)((char*)d_ws + 256);

    init_ws<<<1, 1, 0, stream>>>(amax);
    absmax_kernel<<<64, 256, 0, stream>>>(x, amax);
    quantize_kernel<<<128, 256, 0, stream>>>(x, amax, aws);
    gemm_kernel<<<OUT_F / 128, 256, 0, stream>>>(pw, aws, amax, wscale, bias, out);
}
